// WeatherModel_68504728371284
// MI455X (gfx1250) — matrix operand order repaired and checked by reading, not compiled, not run on hardware
//
#include <hip/hip_runtime.h>
#include <hip/hip_bf16.h>
#include <math.h>
#include <stdint.h>

// ---------------------------------------------------------------------------
// Geometry constants for this model
// ---------------------------------------------------------------------------
#define BB   8
#define TT   10
#define DD   10
#define HH   128
#define WW   128
#define HWSZ (HH * WW)
#define WOUT 5
#define PR   130          // packed rows  (H + halo)
#define PC   130          // packed cols  (W + halo)

typedef __attribute__((ext_vector_type(16))) _Float16 v16h;
typedef __attribute__((ext_vector_type(8)))  _Float16 v8h;
typedef __attribute__((ext_vector_type(8)))  float    v8f;

union V16U { v16h v; v8h h[2]; };
union V8FU { v8f v; float f[8]; };

// A conv input "segment": a run of channels from one tensor, with its own
// channel stride and batch stride.
struct Seg {
    const float* p;
    int  count;
    long cstride;
    long bstride;
};

// ---------------------------------------------------------------------------
// Tensor Data Mover: 1-D contiguous f16 block, global -> LDS.
// D# built per CDNA5 ISA (group0: count/lds/global/type=2,
// group1: data_size=2B, tensor_dim0=tile_dim0=nelem, tile_dim1=1).
// ---------------------------------------------------------------------------
#if __has_builtin(__builtin_amdgcn_tensor_load_to_lds)
#define HAVE_TDM 1
typedef __attribute__((ext_vector_type(4))) unsigned int v4u;
typedef __attribute__((ext_vector_type(8))) int v8i_t;
typedef __attribute__((ext_vector_type(4))) int v4i_t;

__device__ __forceinline__ void tdm_load_1d(unsigned lds_addr,
                                            const _Float16* gptr, int nelem)
{
    unsigned long long ga = (unsigned long long)(uintptr_t)gptr;
    v4u g0;
    g0.x = 1u;                                           // count=1 (valid), flags 0
    g0.y = lds_addr;                                     // LDS byte address
    g0.z = (unsigned)ga;                                 // global_addr[31:0]
    g0.w = (unsigned)((ga >> 32) & 0x01FFFFFFu) | (2u << 30);  // addr[56:32] | type=2
    v8i_t g1;
    g1[0] = (1 << 16);                                   // data_size = 1 -> 2 bytes
    g1[1] = (int)(((unsigned)nelem & 0xFFFFu) << 16);    // tensor_dim0[15:0] @ 63:48
    g1[2] = (int)(((unsigned)nelem >> 16) & 0xFFFFu) | (1 << 16); // dim0 hi | tensor_dim1=1
    g1[3] = (int)(((unsigned)nelem & 0xFFFFu) << 16);    // tile_dim0 = nelem
    g1[4] = 1;                                           // tile_dim1 = 1, tile_dim2 = 0
    g1[5] = nelem;                                       // tensor_dim0_stride[31:0]
    g1[6] = 0;
    g1[7] = 0;
    v4i_t gz = {0, 0, 0, 0};
#if __clang_major__ >= 23
    v8i_t gz8 = {0, 0, 0, 0, 0, 0, 0, 0};
    __builtin_amdgcn_tensor_load_to_lds(g0, g1, gz, gz, gz8, 0);
#else
    __builtin_amdgcn_tensor_load_to_lds(g0, g1, gz, gz, 0);
#endif
}
#endif

// ---------------------------------------------------------------------------
// pack_input: gather concat segments -> f16 packed tensor
//   ipack[nb][r 0..129][c 0..129][ci 0..CiPad)  (zero halo + channel pad)
// This is the exact LDS layout the conv tiles consume, so per-block staging
// becomes a contiguous DMA.
// ---------------------------------------------------------------------------
__global__ __launch_bounds__(256) void pack_input(
    Seg s0, Seg s1, Seg s2, int Ctot, int CiPad, _Float16* __restrict__ ipack)
{
    long idx = (long)blockIdx.x * 256 + threadIdx.x;
    long total = (long)BB * PR * PC * CiPad;
    if (idx >= total) return;
    int ci  = (int)(idx % CiPad);
    long r0 = idx / CiPad;
    int c   = (int)(r0 % PC);
    long r1 = r0 / PC;
    int r   = (int)(r1 % PR);
    int nb  = (int)(r1 / PR);
    int gy = r - 1, gx = c - 1;
    float v = 0.f;
    if ((unsigned)gy < (unsigned)HH && (unsigned)gx < (unsigned)WW && ci < Ctot) {
        int cc = ci;
        const float* p; long cs, bs;
        if (cc < s0.count)              { p = s0.p; cs = s0.cstride; bs = s0.bstride; }
        else { cc -= s0.count;
            if (cc < s1.count)          { p = s1.p; cs = s1.cstride; bs = s1.bstride; }
            else { cc -= s1.count;        p = s2.p; cs = s2.cstride; bs = s2.bstride; } }
        v = p[(long)nb * bs + (long)cc * cs + (long)gy * WW + gx];
    }
    ipack[idx] = (_Float16)v;
}

// pack_weights: OIHW f32 -> f16  wpack[tile][tap][m 0..15][ci 0..CiPad)
__global__ __launch_bounds__(256) void pack_weights(
    const float* __restrict__ w, int Ctot, int CiPad, int Cout,
    _Float16* __restrict__ wpack, long total)
{
    long idx = (long)blockIdx.x * 256 + threadIdx.x;
    if (idx >= total) return;
    int ci   = (int)(idx % CiPad);
    long r0  = idx / CiPad;
    int m    = (int)(r0 % 16);
    long r1  = r0 / 16;
    int tap  = (int)(r1 % 9);
    int tile = (int)(r1 / 9);
    int co   = tile * 16 + m;
    float v = 0.f;
    if (co < Cout && ci < Ctot)
        v = w[((long)co * Ctot + ci) * 9 + tap];
    wpack[idx] = (_Float16)v;
}

// ---------------------------------------------------------------------------
// conv3x3 SAME, NCHW, implicit GEMM on v_wmma_f32_16x16x32_f16.
// Block = 256 threads (8 waves). Each wave -> one 16(co) x 16(px) D tile.
// grid = (H, ceil(Cout/16), B).  mode: 0 = linear, 1 = tanh epilogue.
// Staging: one TDM DMA for the 3-row input tile + one for the weight tile.
// ---------------------------------------------------------------------------
__global__ __launch_bounds__(256) void conv3x3_wmma(
    const _Float16* __restrict__ ipack, const _Float16* __restrict__ wpack,
    int CiPad, const float* __restrict__ bias, int Cout,
    float* __restrict__ out, long obstride, int mode)
{
    extern __shared__ _Float16 smem[];
    _Float16* ilds = smem;                        // [3][130][CiPad]
    _Float16* wlds = smem + 3 * PC * CiPad;       // [9][16][CiPad]

    const int y      = blockIdx.x;
    const int cot    = blockIdx.y;
    const int nb     = blockIdx.z;
    const int cobase = cot * 16;

    const long rowI = (long)PC * CiPad;           // halfs per packed row
    const int  nI   = (int)(3 * rowI);            // input tile halfs
    const int  nW   = 9 * 16 * CiPad;             // weight tile halfs
    const _Float16* gsrcI = ipack + ((long)nb * PR + y) * rowI;  // rows y..y+2
    const _Float16* gsrcW = wpack + (long)cot * nW;

    __builtin_prefetch(gsrcI, 0, 1);              // global_prefetch_b8 warm-up

#if defined(HAVE_TDM)
    if (threadIdx.x == 0) {
        tdm_load_1d((unsigned)(uintptr_t)ilds, gsrcI, nI);
        tdm_load_1d((unsigned)(uintptr_t)wlds, gsrcW, nW);
        __builtin_amdgcn_s_wait_tensorcnt(0);
    }
#else
    {
        const v8h* si = (const v8h*)gsrcI; v8h* di = (v8h*)ilds;
        for (int i = threadIdx.x; i < nI / 8; i += 256) di[i] = si[i];
        const v8h* sw = (const v8h*)gsrcW; v8h* dw = (v8h*)wlds;
        for (int i = threadIdx.x; i < nW / 8; i += 256) dw[i] = sw[i];
    }
#endif
    __syncthreads();

    const int lane = threadIdx.x & 31;
    const int wv   = threadIdx.x >> 5;   // wave id 0..7
    const int x0   = wv * 16;            // pixel tile base in this row
    const int m    = lane & 15;          // A: row M; B: col N
    const int hi   = (lane < 16) ? 0 : 1;

    v8f acc = {};
    const int nchunks = CiPad >> 5;
    for (int tap = 0; tap < 9; ++tap) {
        const int ky = tap / 3, kx = tap % 3;
        const _Float16* wrow = &wlds[(tap * 16 + m) * CiPad];
        const _Float16* irow = &ilds[(ky * PC + x0 + m + kx) * CiPad + hi * 16];
        for (int kc = 0; kc < nchunks; ++kc) {
            V16U a;   // lane<16: K {0..15}; lane>=16: K {16..31}
            a.v = *(const v16h*)(wrow + kc * 32 + hi * 16);
            v16h b = *(const v16h*)(irow + kc * 32);   // lanes hold K-halves
            acc = __builtin_amdgcn_wmma_f32_16x16x32_f16(
                      false, a.v, false, b, (short)0, acc, false, false);
        }
    }

    // epilogue: lane = col N=m, rows M = r + hi*8 ; fuse bias (+ tanh)
    V8FU au; au.v = acc;
    const int xo = x0 + m;
    for (int r = 0; r < 8; ++r) {
        int co = cobase + r + hi * 8;
        if (co < Cout) {
            float v = au.f[r] + bias[co];
            if (mode == 1) v = tanhf(v);
            out[(long)nb * obstride + (long)co * HWSZ + (long)y * WW + xo] = v;
        }
    }
}

// ---------------------------------------------------------------------------
// Pointwise LSTM gate update: z = conv output (B, 4*hid, H, W).
// ---------------------------------------------------------------------------
__device__ __forceinline__ float sigf(float x) { return 1.f / (1.f + __expf(-x)); }

__global__ __launch_bounds__(256) void lstm_pw(
    const float* __restrict__ z,
    const float* __restrict__ cin, long cin_b,
    float* h,  long h_b,
    float* c,  long c_b,
    float* hh, long hh_b,
    float* ch, long ch_b,
    int hid)
{
    int tid = blockIdx.x * 256 + threadIdx.x;
    int per = hid * HWSZ;
    if (tid >= BB * per) return;
    int n  = tid / per;
    int r  = tid % per;
    int cc = r / HWSZ;
    int p  = r % HWSZ;

    long zb = (long)n * 4 * hid * HWSZ;
    float gi = z[zb + (long)cc * HWSZ + p];
    float gf = z[zb + (long)(hid + cc) * HWSZ + p];
    float go = z[zb + (long)(2 * hid + cc) * HWSZ + p];
    float gg = z[zb + (long)(3 * hid + cc) * HWSZ + p];

    float cprev = cin[(long)n * cin_b + r];
    float c2 = sigf(gf) * cprev + sigf(gi) * tanhf(gg);
    float h2 = sigf(go) * tanhf(c2);

    if (h)  h [(long)n * h_b  + r] = h2;
    if (c)  c [(long)n * c_b  + r] = c2;
    if (hh) hh[(long)n * hh_b + r] = h2;
    if (ch) ch[(long)n * ch_b + r] = c2;
}

// Input attention: softmax over D of score, scale all T slices of xx in place.
__global__ __launch_bounds__(256) void attn_scale_xx(
    const float* __restrict__ score, float* __restrict__ xx)
{
    int tid = blockIdx.x * 256 + threadIdx.x;
    if (tid >= BB * HWSZ) return;
    int n = tid / HWSZ, p = tid % HWSZ;
    float s[DD], mx = -1e30f;
    for (int k = 0; k < DD; ++k) {
        s[k] = score[((long)n * DD + k) * HWSZ + p];
        mx = fmaxf(mx, s[k]);
    }
    float sum = 0.f;
    for (int k = 0; k < DD; ++k) { s[k] = __expf(s[k] - mx); sum += s[k]; }
    float inv = 1.f / sum;
    for (int t = 0; t < TT; ++t)
        for (int k = 0; k < DD; ++k)
            xx[(((long)n * TT + t) * DD + k) * HWSZ + p] *= s[k] * inv;
}

// Temporal attention: softmax over T of score -> beta
__global__ __launch_bounds__(256) void softmax_T(
    const float* __restrict__ score, float* __restrict__ beta)
{
    int tid = blockIdx.x * 256 + threadIdx.x;
    if (tid >= BB * HWSZ) return;
    int n = tid / HWSZ, p = tid % HWSZ;
    float s[TT], mx = -1e30f;
    for (int t = 0; t < TT; ++t) {
        s[t] = score[((long)n * TT + t) * HWSZ + p];
        mx = fmaxf(mx, s[t]);
    }
    float sum = 0.f;
    for (int t = 0; t < TT; ++t) { s[t] = __expf(s[t] - mx); sum += s[t]; }
    float inv = 1.f / sum;
    for (int t = 0; t < TT; ++t)
        beta[((long)n * TT + t) * HWSZ + p] = s[t] * inv;
}

// ht = sum_tau beta[:,tau] * hs1[:, ring(tau)]  (64 channels), ring start = s
__global__ __launch_bounds__(256) void weighted_sum64(
    const float* __restrict__ beta, const float* __restrict__ hs1,
    float* __restrict__ ht, int s)
{
    int tid = blockIdx.x * 256 + threadIdx.x;
    int per = 64 * HWSZ;
    if (tid >= BB * per) return;
    int n  = tid / per;
    int r  = tid % per;
    int cc = r / HWSZ;
    int p  = r % HWSZ;
    float acc = 0.f;
    for (int t = 0; t < TT; ++t) {
        int pt = (s + t) % TT;
        acc += beta[((long)n * TT + t) * HWSZ + p] *
               hs1[(((long)n * TT + pt) * 64 + cc) * HWSZ + p];
    }
    ht[(long)n * per + r] = acc;
}

// ---------------------------------------------------------------------------
// Host orchestration
// ---------------------------------------------------------------------------
static inline void launch_conv(hipStream_t st, Seg s0, Seg s1, Seg s2, int Ctot,
                               const float* w, const float* b, int Cout,
                               float* out, long obstride, int mode,
                               _Float16* ipack, _Float16* wpack)
{
    int CiPad  = ((Ctot + 31) / 32) * 32;
    int ntiles = (Cout + 15) / 16;

    long totalI = (long)BB * PR * PC * CiPad;
    pack_input<<<dim3((unsigned)((totalI + 255) / 256)), 256, 0, st>>>(
        s0, s1, s2, Ctot, CiPad, ipack);

    long totalW = (long)ntiles * 9 * 16 * CiPad;
    pack_weights<<<dim3((unsigned)((totalW + 255) / 256)), 256, 0, st>>>(
        w, Ctot, CiPad, Cout, wpack, totalW);

    size_t smem = (size_t)(3 * PC * CiPad + 9 * 16 * CiPad) * sizeof(_Float16);
    dim3 grid(HH, ntiles, BB);
    conv3x3_wmma<<<grid, 256, smem, st>>>(ipack, wpack, CiPad, b, Cout,
                                          out, obstride, mode);
}

extern "C" void kernel_launch(void* const* d_in, const int* in_sizes, int n_in,
                              void* d_out, int out_size, void* d_ws, size_t ws_size,
                              hipStream_t stream)
{
    (void)in_sizes; (void)n_in; (void)out_size; (void)ws_size;

    const float* x      = (const float*)d_in[0];
    const float* h0_in  = (const float*)d_in[1];
    const float* c0_in  = (const float*)d_in[2];
    const float* h1_in  = (const float*)d_in[3];
    const float* c1_in  = (const float*)d_in[4];
    const float* w_enc0 = (const float*)d_in[5];
    const float* b_enc0 = (const float*)d_in[6];
    const float* w_enc1 = (const float*)d_in[7];
    const float* b_enc1 = (const float*)d_in[8];
    const float* w_dec0 = (const float*)d_in[9];
    const float* b_dec0 = (const float*)d_in[10];
    const float* w_dec1 = (const float*)d_in[11];
    const float* b_dec1 = (const float*)d_in[12];
    const float* wa1    = (const float*)d_in[13];
    const float* ba1    = (const float*)d_in[14];
    const float* wa2    = (const float*)d_in[15];
    const float* ba2    = (const float*)d_in[16];
    const float* wt1    = (const float*)d_in[17];
    const float* bt1    = (const float*)d_in[18];
    const float* wt2    = (const float*)d_in[19];
    const float* bt2    = (const float*)d_in[20];

    float* outp = (float*)d_out;   // (B, 5, 1, H, W)

    // workspace layout (floats)
    float* W = (float*)d_ws;
    size_t off = 0;
    float* xx    = W + off; off += (size_t)BB * TT * DD * HWSZ;
    float* h0s   = W + off; off += (size_t)BB * 1  * HWSZ;
    float* c0s   = W + off; off += (size_t)BB * 1  * HWSZ;
    float* h1s   = W + off; off += (size_t)BB * 64 * HWSZ;
    float* c1s   = W + off; off += (size_t)BB * 64 * HWSZ;
    float* hs0   = W + off; off += (size_t)BB * TT * 1  * HWSZ;
    float* cs0   = W + off; off += (size_t)BB * TT * 1  * HWSZ;
    float* hs1   = W + off; off += (size_t)BB * TT * 64 * HWSZ;
    float* cs1   = W + off; off += (size_t)BB * TT * 64 * HWSZ;
    float* zbuf  = W + off; off += (size_t)BB * 256 * HWSZ;
    float* tmp32 = W + off; off += (size_t)BB * 32  * HWSZ;
    float* score = W + off; off += (size_t)BB * TT  * HWSZ;
    float* beta  = W + off; off += (size_t)BB * TT  * HWSZ;
    float* ht64  = W + off; off += (size_t)BB * 64  * HWSZ;
    float* y64   = W + off; off += (size_t)BB * 64  * HWSZ;
    // f16 pack buffers (sized for CiPad max = 160)
    _Float16* ipack = (_Float16*)(W + off); off += (size_t)BB * PR * PC * 160 / 2;
    _Float16* wpack = (_Float16*)(W + off); off += (size_t)16 * 9 * 16 * 160 / 2;

    const Seg SNULL = { nullptr, 0, 0, 0 };

    hipMemcpyAsync(xx,  x,     (size_t)BB*TT*DD*HWSZ*4, hipMemcpyDeviceToDevice, stream);
    hipMemcpyAsync(h0s, h0_in, (size_t)BB*HWSZ*4,       hipMemcpyDeviceToDevice, stream);
    hipMemcpyAsync(c0s, c0_in, (size_t)BB*HWSZ*4,       hipMemcpyDeviceToDevice, stream);
    hipMemcpyAsync(h1s, h1_in, (size_t)BB*64*HWSZ*4,    hipMemcpyDeviceToDevice, stream);
    hipMemcpyAsync(c1s, c1_in, (size_t)BB*64*HWSZ*4,    hipMemcpyDeviceToDevice, stream);

    const int ewBlocks1  = (BB * HWSZ + 255) / 256;
    const int ewBlocks64 = (BB * 64 * HWSZ + 255) / 256;

    // ================= encoder: t = 0..T-1 =================
    for (int t = 0; t < TT; ++t) {
        for (int k = 0; k < DD; ++k) {
            Seg sx = { xx + (long)k * HWSZ, TT, (long)DD * HWSZ, (long)TT * DD * HWSZ };
            Seg sh = { h0s, 1, HWSZ, HWSZ };
            Seg sc = { c0s, 1, HWSZ, HWSZ };
            launch_conv(stream, sx, sh, sc, 12, wa1, ba1, 32,
                        tmp32, (long)32 * HWSZ, 1, ipack, wpack);
            Seg st0 = { tmp32, 32, HWSZ, (long)32 * HWSZ };
            launch_conv(stream, st0, SNULL, SNULL, 32, wa2, ba2, 1,
                        score + (long)k * HWSZ, (long)DD * HWSZ, 0, ipack, wpack);
        }
        attn_scale_xx<<<ewBlocks1, 256, 0, stream>>>(score, xx);

        {   // encoder layer 0 LSTM (in 10, hid 1)
            Seg sx = { xx + (long)t * DD * HWSZ, DD, HWSZ, (long)TT * DD * HWSZ };
            Seg sh = { h0s, 1, HWSZ, HWSZ };
            launch_conv(stream, sx, sh, SNULL, 11, w_enc0, b_enc0, 4,
                        zbuf, (long)4 * HWSZ, 0, ipack, wpack);
            lstm_pw<<<ewBlocks1, 256, 0, stream>>>(
                zbuf, c0s, HWSZ,
                h0s, HWSZ, c0s, HWSZ,
                hs0 + (long)t * HWSZ, (long)TT * HWSZ,
                cs0 + (long)t * HWSZ, (long)TT * HWSZ, 1);
        }
        {   // encoder layer 1 LSTM (in 1, hid 64)
            Seg sx = { hs0 + (long)t * HWSZ, 1, HWSZ, (long)TT * HWSZ };
            Seg sh = { h1s, 64, HWSZ, (long)64 * HWSZ };
            launch_conv(stream, sx, sh, SNULL, 65, w_enc1, b_enc1, 256,
                        zbuf, (long)256 * HWSZ, 0, ipack, wpack);
            lstm_pw<<<ewBlocks64, 256, 0, stream>>>(
                zbuf, c1s, (long)64 * HWSZ,
                h1s, (long)64 * HWSZ, c1s, (long)64 * HWSZ,
                hs1 + (long)t * 64 * HWSZ, (long)TT * 64 * HWSZ,
                cs1 + (long)t * 64 * HWSZ, (long)TT * 64 * HWSZ, 64);
        }
    }

    // ================= decoder: 5 steps, ring buffers =================
    for (int d = 0; d < WOUT; ++d) {
        const int s     = d % TT;
        const int plast = (s + TT - 1) % TT;

        Seg ySeg;
        if (d == 0) ySeg = { x + (long)(TT - 1) * DD * HWSZ, 1, HWSZ, (long)TT * DD * HWSZ };
        else        ySeg = { outp + (long)(d - 1) * HWSZ, 1, HWSZ, (long)WOUT * HWSZ };

        for (int tau = 0; tau < TT; ++tau) {
            int pt = (s + tau) % TT;
            Seg sh = { hs1 + (long)pt * 64 * HWSZ, 64, HWSZ, (long)TT * 64 * HWSZ };
            Seg sc = { cs1 + (long)pt * 64 * HWSZ, 64, HWSZ, (long)TT * 64 * HWSZ };
            launch_conv(stream, ySeg, sh, sc, 129, wt1, bt1, 32,
                        tmp32, (long)32 * HWSZ, 1, ipack, wpack);
            Seg st0 = { tmp32, 32, HWSZ, (long)32 * HWSZ };
            launch_conv(stream, st0, SNULL, SNULL, 32, wt2, bt2, 1,
                        score + (long)tau * HWSZ, (long)TT * HWSZ, 0, ipack, wpack);
        }
        softmax_T<<<ewBlocks1, 256, 0, stream>>>(score, beta);
        weighted_sum64<<<ewBlocks64, 256, 0, stream>>>(beta, hs1, ht64, s);

        {   // decoder layer 0 LSTM (in 1 (y), hid 64); c from cs1[:, -1]
            Seg sh = { ht64, 64, HWSZ, (long)64 * HWSZ };
            launch_conv(stream, ySeg, sh, SNULL, 65, w_dec0, b_dec0, 256,
                        zbuf, (long)256 * HWSZ, 0, ipack, wpack);
            lstm_pw<<<ewBlocks64, 256, 0, stream>>>(
                zbuf, cs1 + (long)plast * 64 * HWSZ, (long)TT * 64 * HWSZ,
                y64, (long)64 * HWSZ, nullptr, 0,
                hs1 + (long)s * 64 * HWSZ, (long)TT * 64 * HWSZ,
                cs1 + (long)s * 64 * HWSZ, (long)TT * 64 * HWSZ, 64);
        }
        {   // decoder layer 1 LSTM (in 64 (y), hid 1); states from hs0/cs0 ring
            Seg sx = { y64, 64, HWSZ, (long)64 * HWSZ };
            Seg sh = { hs0 + (long)plast * HWSZ, 1, HWSZ, (long)TT * HWSZ };
            launch_conv(stream, sx, sh, SNULL, 65, w_dec1, b_dec1, 4,
                        zbuf, (long)4 * HWSZ, 0, ipack, wpack);
            lstm_pw<<<ewBlocks1, 256, 0, stream>>>(
                zbuf, cs0 + (long)plast * HWSZ, (long)TT * HWSZ,
                outp + (long)d * HWSZ, (long)WOUT * HWSZ, nullptr, 0,
                hs0 + (long)s * HWSZ, (long)TT * HWSZ,
                cs0 + (long)s * HWSZ, (long)TT * HWSZ, 1);
        }
    }
}
